// SPELL_HETEROGENEOUS_1563368096105
// MI455X (gfx1250) — matrix operand order repaired and checked by reading, not compiled, not run on hardware
//
#include <hip/hip_runtime.h>

// ---------------------------------------------------------------------------
// Types / WMMA helpers (CDNA5 / gfx1250, wave32)
// ---------------------------------------------------------------------------
typedef __bf16 bf16_t;
typedef __attribute__((ext_vector_type(16))) __bf16 v16bf;
typedef __attribute__((ext_vector_type(8)))  float  v8f;

union Frag {
    v16bf v;
    uint4 q[2];
    unsigned short s[16];
};

__device__ inline unsigned short f2bf(float f) {
    unsigned int u = __float_as_uint(f);
    unsigned int r = ((u >> 16) & 1u) + 0x7FFFu;   // round-to-nearest-even
    return (unsigned short)((u + r) >> 16);
}

__device__ inline v8f v8zero() {
    v8f z;
#pragma unroll
    for (int i = 0; i < 8; i++) z[i] = 0.0f;
    return z;
}

__device__ inline v8f wmma_bf16(v16bf a, v16bf b, v8f c) {
    // D = A(16x32 bf16) * B(32x16 bf16) + C(16x16 f32)
    return __builtin_amdgcn_wmma_f32_16x16x32_bf16(
        /*neg_a=*/false, a, /*neg_b=*/false, b,
        /*c_mod=*/(short)0, c, /*reuse_a=*/false, /*reuse_b=*/false);
}

// B fragment: prepped so each lane reads 16 contiguous bf16 (32 bytes).
__device__ inline v16bf load_bfrag(const unsigned short* base, int fragIdx, int lane) {
    Frag f;
    const uint4* p = (const uint4*)(base + ((size_t)fragIdx * 32 + lane) * 16);
    f.q[0] = p[0];
    f.q[1] = p[1];
    return f.v;
}

// A fragment from a row-major bf16 tile in LDS (row stride in elements).
// Lane<16: K = {0..7} U {16..23} (+kbase); lane>=16: {8..15} U {24..31}.
__device__ inline v16bf load_afrag_lds(const unsigned short* tile, int stride,
                                       int kbase, int lane) {
    int m = lane & 15, half = lane >> 4;
    const unsigned short* p = tile + m * stride + kbase + half * 8;
    Frag f;
    f.q[0] = *(const uint4*)p;
    f.q[1] = *(const uint4*)(p + 16);
    return f.v;
}

// A fragment gathered from one f32 row (row-major K).
__device__ inline v16bf make_afrag_row(const float* row, int kbase, int lane) {
    int k0 = kbase + (lane >> 4) * 8;
    Frag f;
#pragma unroll
    for (int j = 0; j < 8; j++) f.s[j] = f2bf(row[k0 + j]);
#pragma unroll
    for (int j = 0; j < 8; j++) f.s[8 + j] = f2bf(row[k0 + 16 + j]);
    return f.v;
}

// float atomic max via signed-max / unsigned-min trick (works with NaN init).
__device__ inline void atomicMaxF(float* addr, float val) {
    if (val >= 0.0f) atomicMax((int*)addr, __float_as_int(val));
    else             atomicMin((unsigned int*)addr, __float_as_uint(val));
}

// ---------------------------------------------------------------------------
// Weight prep: convert f32 weight matrix into per-lane WMMA B fragments.
// W layout: transposed==0 -> W[k*N+n] ; transposed==1 -> W[n*K+k]
// ---------------------------------------------------------------------------
__global__ void k_prep(const float* __restrict__ W, unsigned short* __restrict__ dst,
                       int K, int N, int transposed) {
    int t = blockIdx.x * blockDim.x + threadIdx.x;
    if (t >= K * N) return;
    int j       = t & 15;
    int lane    = (t >> 4) & 31;
    int fragIdx = t >> 9;
    int ntiles  = N >> 4;
    int ntile   = fragIdx % ntiles;
    int kstep   = fragIdx / ntiles;
    int half    = lane >> 4;
    int k = kstep * 32 + ((j < 8) ? (half * 8 + j) : (16 + half * 8 + (j - 8)));
    int n = ntile * 16 + (lane & 15);
    float v = transposed ? W[(size_t)n * K + k] : W[(size_t)k * N + n];
    dst[t] = f2bf(v);
}

// ---------------------------------------------------------------------------
// h = relu(bn0(x @ w_lin + b_lin)); x:[N,1024] -> h:[N,64]
// One wave per 16-node tile; 32 K-steps x 4 N-tiles of WMMA.
// ---------------------------------------------------------------------------
__global__ void k_lin(const float* __restrict__ x, const unsigned short* __restrict__ wf,
                      const float* __restrict__ bias,
                      const float* __restrict__ g, const float* __restrict__ be,
                      const float* __restrict__ mu, const float* __restrict__ var,
                      float* __restrict__ h, int nNodes, int inDim) {
    int wv = threadIdx.x >> 5, lane = threadIdx.x & 31;
    int tile = blockIdx.x * 4 + wv;
    if (tile * 16 >= nNodes) return;
    int n0 = tile * 16, m = lane & 15, half = lane >> 4, nloc = lane & 15;
    const float* row = x + (size_t)(n0 + m) * inDim;

    v8f acc[4] = {v8zero(), v8zero(), v8zero(), v8zero()};
    int ksteps = inDim >> 5;
    for (int ks = 0; ks < ksteps; ks++) {
        if (ks + 4 < ksteps) __builtin_prefetch(row + (ks + 4) * 32, 0, 3);
        v16bf a = make_afrag_row(row, ks * 32, lane);
#pragma unroll
        for (int nt = 0; nt < 4; nt++)
            acc[nt] = wmma_bf16(a, load_bfrag(wf, ks * 4 + nt, lane), acc[nt]);
    }
#pragma unroll
    for (int nt = 0; nt < 4; nt++) {
        int n = nt * 16 + nloc;
        float scale = g[n] * rsqrtf(var[n] + 1e-5f);
        float off   = (bias[n] - mu[n]) * scale + be[n];
#pragma unroll
        for (int r = 0; r < 8; r++) {
            float v = acc[nt][r] * scale + off;
            h[(size_t)(n0 + half * 8 + r) * 64 + n] = v > 0.0f ? v : 0.0f;
        }
    }
}

// ---------------------------------------------------------------------------
// EdgeConv: m = relu([xi, xj-xi] @ w1 + b1) @ w2 + b2 ; atomic segment-max.
// gridDim.y selects which of the 3 EdgeConvs (mask rule differs).
// ---------------------------------------------------------------------------
struct ECParams {
    const unsigned short* w1f[3];
    const unsigned short* w2f[3];
    const float* b1[3];
    const float* b2[3];
    float* agg[3];
};

__global__ void k_edge(const float* __restrict__ h,
                       const int* __restrict__ srcI, const int* __restrict__ dstI,
                       const int* __restrict__ attr, ECParams P, int nEdges) {
    __shared__ __align__(16) unsigned short lA[4][16 * 128];
    __shared__ int sDst[4][16];
    __shared__ int sAttr[4][16];
    __shared__ int sSrc[4][16];

    int wv = threadIdx.x >> 5, lane = threadIdx.x & 31;
    int tile = blockIdx.x * 4 + wv;
    int conv = blockIdx.y;
    int e0 = tile * 16;
    bool active = (e0 < nEdges);

    if (active && lane < 16) {
        int e = e0 + lane;
        sDst[wv][lane]  = dstI[e];
        sSrc[wv][lane]  = srcI[e];
        sAttr[wv][lane] = attr[e];
    }
    __syncthreads();

    if (active) {                        // build A tile [16 x 128] = [xi | xj-xi]
        for (int e = 0; e < 16; e++) {
            const float* xi = h + (size_t)sDst[wv][e] * 64;
            const float* xj = h + (size_t)sSrc[wv][e] * 64;
            for (int c = lane; c < 64; c += 32) {
                float vi = xi[c], vj = xj[c];
                lA[wv][e * 128 + c]      = f2bf(vi);
                lA[wv][e * 128 + 64 + c] = f2bf(vj - vi);
            }
        }
    }
    __syncthreads();

    v8f acc[4] = {v8zero(), v8zero(), v8zero(), v8zero()};
    if (active) {
#pragma unroll
        for (int ks = 0; ks < 4; ks++) {
            v16bf a = load_afrag_lds(lA[wv], 128, ks * 32, lane);
#pragma unroll
            for (int nt = 0; nt < 4; nt++)
                acc[nt] = wmma_bf16(a, load_bfrag(P.w1f[conv], ks * 4 + nt, lane), acc[nt]);
        }
    }
    __syncthreads();

    int half = lane >> 4, nloc = lane & 15;
    if (active) {                        // relu(+b1) -> bf16 LDS tile [16 x 64]
#pragma unroll
        for (int nt = 0; nt < 4; nt++) {
            int n = nt * 16 + nloc;
            float bb = P.b1[conv][n];
#pragma unroll
            for (int r = 0; r < 8; r++) {
                float v = acc[nt][r] + bb;
                lA[wv][(half * 8 + r) * 64 + n] = f2bf(v > 0.0f ? v : 0.0f);
            }
        }
    }
    __syncthreads();

    if (active) {
        v8f a2[4] = {v8zero(), v8zero(), v8zero(), v8zero()};
#pragma unroll
        for (int ks = 0; ks < 2; ks++) {
            v16bf a = load_afrag_lds(lA[wv], 64, ks * 32, lane);
#pragma unroll
            for (int nt = 0; nt < 4; nt++)
                a2[nt] = wmma_bf16(a, load_bfrag(P.w2f[conv], ks * 4 + nt, lane), a2[nt]);
        }
        // masked scatter-max to agg[dst]
#pragma unroll
        for (int nt = 0; nt < 4; nt++) {
            int n = nt * 16 + nloc;
            float bb = P.b2[conv][n];
#pragma unroll
            for (int r = 0; r < 8; r++) {
                int m = half * 8 + r;
                int a = sAttr[wv][m];
                bool ok = (conv == 0) ? (a <= 0) : (conv == 1) ? (a >= 0) : true;
                if (ok) {
                    float v = a2[nt][r] + bb;
                    atomicMaxF(&P.agg[conv][(size_t)sDst[wv][m] * 64 + n], v);
                }
            }
        }
    }
}

// ---------------------------------------------------------------------------
// agg -> (non-finite -> 0) -> BN -> relu   (in place), gridDim.y = conv
// ---------------------------------------------------------------------------
struct BNParams {
    const float* g[3]; const float* be[3]; const float* mu[3]; const float* var[3];
    float* agg[3];
};

__global__ void k_bnrelu(BNParams P, int total) {
    int conv = blockIdx.y;
    int idx = blockIdx.x * blockDim.x + threadIdx.x;
    if (idx >= total) return;
    int ch = idx & 63;
    float v = P.agg[conv][idx];
    unsigned bits = __float_as_uint(v);
    if ((bits & 0x7F800000u) == 0x7F800000u) v = 0.0f;   // inf/nan -> 0
    float scale = P.g[conv][ch] * rsqrtf(P.var[conv][ch] + 1e-5f);
    v = (v - P.mu[conv][ch]) * scale + P.be[conv][ch];
    P.agg[conv][idx] = v > 0.0f ? v : 0.0f;
}

// ---------------------------------------------------------------------------
// RGCN projection: base = sum_c x_c@root_c (+sum bias); p[c][r] = x_c@wrel_c[r]
// ---------------------------------------------------------------------------
struct RGParams {
    const unsigned short* rootf[3];
    const unsigned short* wrelf[3][2];
    const float* bias[3];
    const float* xin[3];
    float* p[3][2];
};

__global__ void k_rgcn(RGParams P, float* __restrict__ outbuf, int nNodes) {
    int wv = threadIdx.x >> 5, lane = threadIdx.x & 31;
    int tile = blockIdx.x * 4 + wv;
    if (tile * 16 >= nNodes) return;
    int n0 = tile * 16, m = lane & 15, half = lane >> 4, nloc = lane & 15;

    v8f base[2] = {v8zero(), v8zero()};
    v8f pr[3][2][2];
#pragma unroll
    for (int c = 0; c < 3; c++)
#pragma unroll
        for (int r = 0; r < 2; r++)
#pragma unroll
            for (int nt = 0; nt < 2; nt++) pr[c][r][nt] = v8zero();

#pragma unroll
    for (int c = 0; c < 3; c++) {
        const float* row = P.xin[c] + (size_t)(n0 + m) * 64;
#pragma unroll
        for (int ks = 0; ks < 2; ks++) {
            v16bf a = make_afrag_row(row, ks * 32, lane);
#pragma unroll
            for (int nt = 0; nt < 2; nt++) {
                base[nt]     = wmma_bf16(a, load_bfrag(P.rootf[c],    ks * 2 + nt, lane), base[nt]);
                pr[c][0][nt] = wmma_bf16(a, load_bfrag(P.wrelf[c][0], ks * 2 + nt, lane), pr[c][0][nt]);
                pr[c][1][nt] = wmma_bf16(a, load_bfrag(P.wrelf[c][1], ks * 2 + nt, lane), pr[c][1][nt]);
            }
        }
    }
#pragma unroll
    for (int nt = 0; nt < 2; nt++) {
        int n = nt * 16 + nloc;
        float bb = P.bias[0][n] + P.bias[1][n] + P.bias[2][n];
#pragma unroll
        for (int r = 0; r < 8; r++) {
            size_t nd = (size_t)(n0 + half * 8 + r);
            outbuf[nd * 32 + n] = base[nt][r] + bb;
#pragma unroll
            for (int c = 0; c < 3; c++) {
                P.p[c][0][nd * 32 + n] = pr[c][0][nt][r];
                P.p[c][1][nd * 32 + n] = pr[c][1][nt][r];
            }
        }
    }
}

// ---------------------------------------------------------------------------
// RGCN scatter: sum[c][r][dst] += p[c][r][src]; cnt[c][r][dst] += 1
// 4 threads/edge x 8 channels. gridDim.y = conv.
// ---------------------------------------------------------------------------
struct SCParams {
    const float* p[3][2];
    float* sum[3][2];
    float* cnt[3][2];
    const int* src; const int* dst; const int* attr;
};

__global__ void k_scatter(SCParams P, int nEdges) {
    int conv = blockIdx.y;
    int idx = blockIdx.x * blockDim.x + threadIdx.x;
    int e = idx >> 2, quad = idx & 3;
    if (e >= nEdges) return;
    int a = P.attr[e];
    bool ok = (conv == 0) ? (a <= 0) : (conv == 1) ? (a >= 0) : true;
    if (!ok) return;
    int r = (a != -2) ? 1 : 0;
    int s = P.src[e], d = P.dst[e];
    const float* ps = P.p[conv][r] + (size_t)s * 32 + quad * 8;
    float* sd = P.sum[conv][r] + (size_t)d * 32 + quad * 8;
#pragma unroll
    for (int j = 0; j < 8; j++) atomicAdd(sd + j, ps[j]);
    if (quad == 0) atomicAdd(P.cnt[conv][r] + d, 1.0f);
}

// ---------------------------------------------------------------------------
// finalize: out = base + sum/max(cnt,1); write d_out slot 0 + transposed copy
// ---------------------------------------------------------------------------
struct FZParams { const float* sum[3][2]; const float* cnt[3][2]; };

__global__ void k_finalize(FZParams P, const float* __restrict__ outbuf,
                           float* __restrict__ dout0, float* __restrict__ outT,
                           int nNodes) {
    int idx = blockIdx.x * blockDim.x + threadIdx.x;
    if (idx >= nNodes * 32) return;
    int node = idx >> 5, ch = idx & 31;
    float v = outbuf[idx];
#pragma unroll
    for (int c = 0; c < 3; c++)
#pragma unroll
        for (int r = 0; r < 2; r++)
            v += P.sum[c][r][idx] / fmaxf(P.cnt[c][r][node], 1.0f);
    dout0[idx] = v;
    outT[(size_t)ch * nNodes + node] = v;
}

// ---------------------------------------------------------------------------
// softmax over 32 channels (channel-major [32][T])
// ---------------------------------------------------------------------------
__global__ void k_softmax(const float* __restrict__ inT, float* __restrict__ outT, int T) {
    int t = blockIdx.x * blockDim.x + threadIdx.x;
    if (t >= T) return;
    float mx = -3.4e38f;
#pragma unroll
    for (int c = 0; c < 32; c++) mx = fmaxf(mx, inT[(size_t)c * T + t]);
    float e[32], s = 0.0f;
#pragma unroll
    for (int c = 0; c < 32; c++) { e[c] = __expf(inT[(size_t)c * T + t] - mx); s += e[c]; }
    float inv = 1.0f / s;
#pragma unroll
    for (int c = 0; c < 32; c++) outT[(size_t)c * T + t] = e[c] * inv;
}

// ---------------------------------------------------------------------------
// f = W_in(64x32) @ sm + b_in   (1x1 conv; K=32, N=64)
// ---------------------------------------------------------------------------
__global__ void k_conv_in(const float* __restrict__ smT, const unsigned short* __restrict__ wf,
                          const float* __restrict__ b, float* __restrict__ f, int T) {
    int wv = threadIdx.x >> 5, lane = threadIdx.x & 31;
    int tile = blockIdx.x * 4 + wv;
    if (tile * 16 >= T) return;
    int t0 = tile * 16, m = lane & 15, half = lane >> 4, nloc = lane & 15;
    int t = t0 + m;
    Frag fa;
#pragma unroll
    for (int j = 0; j < 8; j++) fa.s[j]     = f2bf(smT[(size_t)(half * 8 + j) * T + t]);
#pragma unroll
    for (int j = 0; j < 8; j++) fa.s[8 + j] = f2bf(smT[(size_t)(16 + half * 8 + j) * T + t]);
    v8f acc[4] = {v8zero(), v8zero(), v8zero(), v8zero()};
#pragma unroll
    for (int nt = 0; nt < 4; nt++)
        acc[nt] = wmma_bf16(fa.v, load_bfrag(wf, nt, lane), acc[nt]);
#pragma unroll
    for (int nt = 0; nt < 4; nt++) {
        int n = nt * 16 + nloc;
        float bb = b[n];
#pragma unroll
        for (int r = 0; r < 8; r++)
            f[(size_t)n * T + t0 + half * 8 + r] = acc[nt][r] + bb;
    }
}

// ---------------------------------------------------------------------------
// dilated residual layer: fout = fin + W1 @ relu(Wd *d fin + bd) + b1
// GEMM1: K=192 (cin*3+tap), GEMM2: K=64 via LDS bf16 tile.
// ---------------------------------------------------------------------------
__global__ void k_dilated(const float* __restrict__ fin, float* __restrict__ fout,
                          const unsigned short* __restrict__ wdf, const float* __restrict__ bd,
                          const unsigned short* __restrict__ w1f, const float* __restrict__ b1,
                          int dil, int T) {
    __shared__ __align__(16) unsigned short lM[4][16 * 64];
    int wv = threadIdx.x >> 5, lane = threadIdx.x & 31;
    int tile = blockIdx.x * 4 + wv;
    bool active = (tile * 16 < T);
    int t0 = tile * 16, m = lane & 15, half = lane >> 4, nloc = lane & 15;
    int t = t0 + m;

    v8f acc[4] = {v8zero(), v8zero(), v8zero(), v8zero()};
    if (active) {
        for (int ks = 0; ks < 6; ks++) {
            Frag fa;
#pragma unroll
            for (int jj = 0; jj < 16; jj++) {
                int koff = (jj < 8) ? (half * 8 + jj) : (16 + half * 8 + (jj - 8));
                int k = ks * 32 + koff;
                int cin = k / 3;
                int kw = k - cin * 3;
                int tt = t + (kw - 1) * dil;
                float v = (tt >= 0 && tt < T) ? fin[(size_t)cin * T + tt] : 0.0f;
                fa.s[jj] = f2bf(v);
            }
#pragma unroll
            for (int nt = 0; nt < 4; nt++)
                acc[nt] = wmma_bf16(fa.v, load_bfrag(wdf, ks * 4 + nt, lane), acc[nt]);
        }
    }
    __syncthreads();
    if (active) {
#pragma unroll
        for (int nt = 0; nt < 4; nt++) {
            int n = nt * 16 + nloc;
            float bb = bd[n];
#pragma unroll
            for (int r = 0; r < 8; r++) {
                float v = acc[nt][r] + bb;
                lM[wv][(half * 8 + r) * 64 + n] = f2bf(v > 0.0f ? v : 0.0f);
            }
        }
    }
    __syncthreads();
    if (active) {
        v8f a2[4] = {v8zero(), v8zero(), v8zero(), v8zero()};
#pragma unroll
        for (int ks = 0; ks < 2; ks++) {
            v16bf a = load_afrag_lds(lM[wv], 64, ks * 32, lane);
#pragma unroll
            for (int nt = 0; nt < 4; nt++)
                a2[nt] = wmma_bf16(a, load_bfrag(w1f, ks * 4 + nt, lane), a2[nt]);
        }
#pragma unroll
        for (int nt = 0; nt < 4; nt++) {
            int n = nt * 16 + nloc;
            float bb = b1[n];
#pragma unroll
            for (int r = 0; r < 8; r++) {
                size_t tt = (size_t)t0 + half * 8 + r;
                fout[(size_t)n * T + tt] = fin[(size_t)n * T + tt] + a2[nt][r] + bb;
            }
        }
    }
}

// ---------------------------------------------------------------------------
// conv_out: xr = W_out(32x64) @ f + b_out; writes channel-major copy + d_out slice
// ---------------------------------------------------------------------------
__global__ void k_conv_out(const float* __restrict__ f, const unsigned short* __restrict__ wf,
                           const float* __restrict__ b, float* __restrict__ xrT,
                           float* __restrict__ doutSlice, int T) {
    int wv = threadIdx.x >> 5, lane = threadIdx.x & 31;
    int tile = blockIdx.x * 4 + wv;
    if (tile * 16 >= T) return;
    int t0 = tile * 16, m = lane & 15, half = lane >> 4, nloc = lane & 15;
    int t = t0 + m;
    v8f acc[2] = {v8zero(), v8zero()};
#pragma unroll
    for (int ks = 0; ks < 2; ks++) {
        Frag fa;
#pragma unroll
        for (int j = 0; j < 8; j++) fa.s[j]     = f2bf(f[(size_t)(ks * 32 + half * 8 + j) * T + t]);
#pragma unroll
        for (int j = 0; j < 8; j++) fa.s[8 + j] = f2bf(f[(size_t)(ks * 32 + 16 + half * 8 + j) * T + t]);
#pragma unroll
        for (int nt = 0; nt < 2; nt++)
            acc[nt] = wmma_bf16(fa.v, load_bfrag(wf, ks * 2 + nt, lane), acc[nt]);
    }
#pragma unroll
    for (int nt = 0; nt < 2; nt++) {
        int n = nt * 16 + nloc;
        float bb = b[n];
#pragma unroll
        for (int r = 0; r < 8; r++) {
            int tt = t0 + half * 8 + r;
            float v = acc[nt][r] + bb;
            xrT[(size_t)n * T + tt] = v;
            doutSlice[(size_t)tt * 32 + n] = v;
        }
    }
}

// ---------------------------------------------------------------------------
// Host driver
// ---------------------------------------------------------------------------
extern "C" void kernel_launch(void* const* d_in, const int* in_sizes, int n_in,
                              void* d_out, int out_size, void* d_ws, size_t ws_size,
                              hipStream_t stream) {
    (void)in_sizes; (void)n_in; (void)out_size; (void)ws_size;
    const int N = 60000, E = 960000, IN = 1024, T = N;

    auto F = [&](int i) { return (const float*)d_in[i]; };
    const float* x    = F(0);
    const int*   src  = (const int*)d_in[1];
    const int*   dst  = (const int*)d_in[1] + E;
    const int*   attr = (const int*)d_in[2];
    // flattened param indices (setup_inputs insertion order):
    // 3:w_lin 4:b_lin; bn_k = 5+4k (g,b,m,v); ec_c = 21+4c (w1,b1,w2,b2);
    // rg_c = 33+3c (root,wrel,bias); ref_s = 42+44s (w_in,b_in,10x[wd,bd,w1,b1],w_out,b_out)

    char* wp = (char*)d_ws;
    auto alloc = [&](size_t bytes) -> void* {
        void* p = (void*)wp;
        wp += (bytes + 255) & ~(size_t)255;
        return p;
    };
    float* h = (float*)alloc((size_t)N * 64 * 4);
    float* agg[3];
    for (int c = 0; c < 3; c++) agg[c] = (float*)alloc((size_t)N * 64 * 4);
    float* pbuf[3][2]; float* sum[3][2]; float* cnt[3][2];
    for (int c = 0; c < 3; c++)
        for (int r = 0; r < 2; r++) {
            pbuf[c][r] = (float*)alloc((size_t)N * 32 * 4);
            sum[c][r]  = (float*)alloc((size_t)N * 32 * 4);
            cnt[c][r]  = (float*)alloc((size_t)N * 4);
        }
    float* outbuf = (float*)alloc((size_t)N * 32 * 4);
    float* outT   = (float*)alloc((size_t)32 * T * 4);
    float* smT    = (float*)alloc((size_t)32 * T * 4);
    float* fA     = (float*)alloc((size_t)64 * T * 4);
    float* fB     = (float*)alloc((size_t)64 * T * 4);
    float* xrT1   = (float*)alloc((size_t)32 * T * 4);

    unsigned short* fr_wlin = (unsigned short*)alloc((size_t)IN * 64 * 2);
    unsigned short *fr_ecw1[3], *fr_ecw2[3], *fr_root[3], *fr_wrel[3][2];
    for (int c = 0; c < 3; c++) {
        fr_ecw1[c] = (unsigned short*)alloc(128 * 64 * 2);
        fr_ecw2[c] = (unsigned short*)alloc(64 * 64 * 2);
        fr_root[c] = (unsigned short*)alloc(64 * 32 * 2);
        fr_wrel[c][0] = (unsigned short*)alloc(64 * 32 * 2);
        fr_wrel[c][1] = (unsigned short*)alloc(64 * 32 * 2);
    }
    unsigned short *fr_win[3], *fr_wout[3], *fr_wd[3][10], *fr_w1[3][10];
    for (int s = 0; s < 3; s++) {
        fr_win[s]  = (unsigned short*)alloc(32 * 64 * 2);
        fr_wout[s] = (unsigned short*)alloc(64 * 32 * 2);
        for (int l = 0; l < 10; l++) {
            fr_wd[s][l] = (unsigned short*)alloc(192 * 64 * 2);
            fr_w1[s][l] = (unsigned short*)alloc(64 * 64 * 2);
        }
    }

    auto prep = [&](const float* W, unsigned short* dp, int K, int Nc, int tr) {
        int tot = K * Nc;
        k_prep<<<dim3((tot + 255) / 256), dim3(256), 0, stream>>>(W, dp, K, Nc, tr);
    };
    prep(F(3), fr_wlin, 1024, 64, 0);
    for (int c = 0; c < 3; c++) {
        prep(F(21 + c * 4 + 0), fr_ecw1[c], 128, 64, 0);
        prep(F(21 + c * 4 + 2), fr_ecw2[c], 64, 64, 0);
        prep(F(33 + c * 3 + 0), fr_root[c], 64, 32, 0);
        prep(F(33 + c * 3 + 1),            fr_wrel[c][0], 64, 32, 0);
        prep(F(33 + c * 3 + 1) + 64 * 32,  fr_wrel[c][1], 64, 32, 0);
    }
    for (int s = 0; s < 3; s++) {
        int rb = 42 + s * 44;
        prep(F(rb + 0), fr_win[s], 32, 64, 1);                 // w_in: [64][32] -> B^T
        for (int l = 0; l < 10; l++) {
            prep(F(rb + 2 + l * 4 + 0), fr_wd[s][l], 192, 64, 1);
            prep(F(rb + 2 + l * 4 + 2), fr_w1[s][l], 64, 64, 1);
        }
        prep(F(rb + 42), fr_wout[s], 64, 32, 1);
    }

    // init aggregation buffers: agg -> NaN bits (non-finite), sum/cnt -> 0
    for (int c = 0; c < 3; c++)
        hipMemsetAsync(agg[c], 0xFF, (size_t)N * 64 * 4, stream);
    for (int c = 0; c < 3; c++)
        for (int r = 0; r < 2; r++) {
            hipMemsetAsync(sum[c][r], 0, (size_t)N * 32 * 4, stream);
            hipMemsetAsync(cnt[c][r], 0, (size_t)N * 4, stream);
        }

    // 1) h = relu(bn0(x @ w_lin + b))
    k_lin<<<dim3((N / 16 + 3) / 4), dim3(128), 0, stream>>>(
        x, fr_wlin, F(4), F(5), F(6), F(7), F(8), h, N, IN);

    // 2) three EdgeConvs (gridDim.y selects mask + params)
    ECParams ec{};
    for (int c = 0; c < 3; c++) {
        ec.w1f[c] = fr_ecw1[c]; ec.w2f[c] = fr_ecw2[c];
        ec.b1[c] = F(21 + c * 4 + 1); ec.b2[c] = F(21 + c * 4 + 3);
        ec.agg[c] = agg[c];
    }
    k_edge<<<dim3(E / 16 / 4, 3), dim3(128), 0, stream>>>(h, src, dst, attr, ec, E);

    // 3) bn+relu (non-finite -> 0)
    BNParams bn{};
    for (int c = 0; c < 3; c++) {
        bn.g[c] = F(9 + c * 4 + 0); bn.be[c] = F(9 + c * 4 + 1);
        bn.mu[c] = F(9 + c * 4 + 2); bn.var[c] = F(9 + c * 4 + 3);
        bn.agg[c] = agg[c];
    }
    k_bnrelu<<<dim3((N * 64 + 255) / 256, 3), dim3(256), 0, stream>>>(bn, N * 64);

    // 4) RGCN project + scatter + finalize
    RGParams rg{};
    for (int c = 0; c < 3; c++) {
        rg.rootf[c] = fr_root[c];
        rg.wrelf[c][0] = fr_wrel[c][0]; rg.wrelf[c][1] = fr_wrel[c][1];
        rg.bias[c] = F(33 + c * 3 + 2);
        rg.xin[c] = agg[c];
        rg.p[c][0] = pbuf[c][0]; rg.p[c][1] = pbuf[c][1];
    }
    k_rgcn<<<dim3((N / 16 + 3) / 4), dim3(128), 0, stream>>>(rg, outbuf, N);

    SCParams sc{};
    for (int c = 0; c < 3; c++)
        for (int r = 0; r < 2; r++) {
            sc.p[c][r] = pbuf[c][r]; sc.sum[c][r] = sum[c][r]; sc.cnt[c][r] = cnt[c][r];
        }
    sc.src = src; sc.dst = dst; sc.attr = attr;
    k_scatter<<<dim3((E * 4 + 255) / 256, 3), dim3(256), 0, stream>>>(sc, E);

    FZParams fz{};
    for (int c = 0; c < 3; c++)
        for (int r = 0; r < 2; r++) { fz.sum[c][r] = sum[c][r]; fz.cnt[c][r] = cnt[c][r]; }
    k_finalize<<<dim3((N * 32 + 255) / 256), dim3(256), 0, stream>>>(
        fz, outbuf, (float*)d_out, outT, N);

    // 5) three MS-TCN refinement stages
    for (int s = 0; s < 3; s++) {
        int rb = 42 + s * 44;
        const float* in = (s == 1) ? xrT1 : outT;   // s0: outT, s1: xrT1, s2: outT
        float* outp     = (s == 1) ? outT : xrT1;
        k_softmax<<<dim3((T + 255) / 256), dim3(256), 0, stream>>>(in, smT, T);
        k_conv_in<<<dim3((T / 16 + 3) / 4), dim3(128), 0, stream>>>(
            smT, fr_win[s], F(rb + 1), fA, T);
        for (int l = 0; l < 10; l++) {
            const float* fi = (l & 1) ? fB : fA;
            float* fo       = (l & 1) ? fA : fB;
            k_dilated<<<dim3((T / 16 + 3) / 4), dim3(128), 0, stream>>>(
                fi, fo, fr_wd[s][l], F(rb + 2 + l * 4 + 1),
                fr_w1[s][l], F(rb + 2 + l * 4 + 3), 1 << l, T);
        }
        k_conv_out<<<dim3((T / 16 + 3) / 4), dim3(128), 0, stream>>>(
            fA, fr_wout[s], F(rb + 43), outp,
            (float*)d_out + (size_t)(s + 1) * N * 32, T);
    }
}
